// AttentionCell_6270652252666
// MI455X (gfx1250) — matrix operand order repaired and checked by reading, not compiled, not run on hardware
//
#include <hip/hip_runtime.h>
#include <hip/hip_bf16.h>
#include <math.h>

// Problem sizes (fixed by the reference)
#define B_SZ   128
#define NIN    512
#define NHID   1024
#define LCTX   512
#define DCTX   1024

typedef __attribute__((ext_vector_type(16))) _Float16 v16h;
typedef __attribute__((ext_vector_type(8)))  float    v8f;

// Fast activation helpers: one v_rcp_f32 instead of an IEEE-division chain.
__device__ __forceinline__ float sigm(float x) {
    return __builtin_amdgcn_rcpf(1.0f + __expf(-x));
}
__device__ __forceinline__ float tanh_fast(float x) {
    // tanh(x) = 1 - 2/(exp(2x)+1); saturates correctly at +-1 for large |x|
    const float e = __expf(2.0f * x);
    return 1.0f - 2.0f * __builtin_amdgcn_rcpf(e + 1.0f);
}

// ---------------------------------------------------------------------------
// Mixed-precision WMMA GEMM:  C[M,N] = epi( concat(A0,A1)[M,K] @ Bw[K,N] )
//   A(b,k) = k < split ? A0[b*lda0+k] : A1[b*lda1+(k-split)]
//   EPI: 0 = identity, 1 = sigmoid(v + bias[col]), 2 = tanh(v)
// Workgroup: 256 threads = 8 waves; WG tile 64(M) x 64(N); K staged in LDS
// 64 wide. Waves laid out 2(M-groups of 32) x 4(N cols of 16); each wave owns
// two M-stacked 16x16 tiles sharing one B fragment -> 4 v_wmma per stage.
// M = 128, K % 64 == 0, N % 64 == 0, split % 64 == 0 for every call here.
// ---------------------------------------------------------------------------
template<int EPI>
__global__ __launch_bounds__(256)
void gemm_wmma(const float* __restrict__ A0, int lda0, int split,
               const float* __restrict__ A1, int lda1,
               const float* __restrict__ Bw, const float* __restrict__ bias,
               float* __restrict__ C, int N, int K)
{
    __shared__ _Float16 As[64][72];   // 64 M rows x 64 K (padded)
    __shared__ _Float16 Bs[64][72];   // 64 K rows x 64 N (padded)

    const int tid  = threadIdx.x;
    const int lane = tid & 31;
    const int wave = tid >> 5;
    const int wm   = wave & 1;        // M group (0..1) -> rows wm*32 .. +31
    const int wn   = wave >> 1;       // N col   (0..3) -> cols wn*16 .. +15
    const int bm   = blockIdx.x * 64;
    const int bn   = blockIdx.y * 64;

    // Staging: 16 consecutive elements per thread (A and B each)
    const int sr = tid >> 2;          // 0..63 row
    const int sc = (tid & 3) * 16;    // col start

    v8f acc0 = {};
    v8f acc1 = {};

    for (int k0 = 0; k0 < K; k0 += 64) {
        if (k0 + 64 < K) {
            __builtin_prefetch(&Bw[(size_t)(k0 + 64 + sr) * N + bn + sc], 0, 3);
        }
        // ---- stage A tile (virtual concat A0|A1), f32 -> f16 ----
        {
            const int row = bm + sr;
            #pragma unroll
            for (int i = 0; i < 16; ++i) {
                const int kg = k0 + sc + i;
                const float v = (kg < split)
                    ? A0[(size_t)row * lda0 + kg]
                    : A1[(size_t)row * lda1 + (kg - split)];
                As[sr][sc + i] = (_Float16)v;
            }
        }
        // ---- stage B tile ----
        {
            const float* bp = &Bw[(size_t)(k0 + sr) * N + bn + sc];
            #pragma unroll
            for (int i = 0; i < 16; ++i) Bs[sr][sc + i] = (_Float16)bp[i];
        }
        __syncthreads();

        // ---- 2 K-substeps of 32, 2 M-subtiles each: 4 WMMAs per stage ----
        const int h  = lane >> 4;                 // lane half
        const int ml = lane & 15;                 // row/col within 16
        #pragma unroll
        for (int kk = 0; kk < 64; kk += 32) {
            // B fragment: lane -> N, elems -> K (table order, same as A)
            v16h bfrag;
            const int ncol = wn * 16 + ml;
            #pragma unroll
            for (int e = 0; e < 8; ++e) {
                bfrag[e]     = Bs[kk + h * 8 + e][ncol];
                bfrag[e + 8] = Bs[kk + 16 + h * 8 + e][ncol];
            }

            // A fragments for the two M subtiles (rows wm*32 + {0,16} + ml)
            v16h a0, a1;
            const int r0 = wm * 32 + ml;
            const int r1 = r0 + 16;
            #pragma unroll
            for (int e = 0; e < 8; ++e) {
                a0[e]     = As[r0][kk + h * 8 + e];
                a0[e + 8] = As[r0][kk + 16 + h * 8 + e];
                a1[e]     = As[r1][kk + h * 8 + e];
                a1[e + 8] = As[r1][kk + 16 + h * 8 + e];
            }

            acc0 = __builtin_amdgcn_wmma_f32_16x16x32_f16(
                       false, a0, false, bfrag, (short)0, acc0, false, false);
            acc1 = __builtin_amdgcn_wmma_f32_16x16x32_f16(
                       false, a1, false, bfrag, (short)0, acc1, false, false);
        }
        __syncthreads();
    }

    // ---- epilogue: C/D layout -> VGPR i holds M = i + 8*(lane>=16), lane -> N
    const int n = lane & 15;
    const int g = lane >> 4;
    const int col = bn + wn * 16 + n;
    const float bv = (EPI == 1) ? bias[col] : 0.0f;
    #pragma unroll
    for (int i = 0; i < 8; ++i) {
        const int row0 = bm + wm * 32 + g * 8 + i;
        float v0 = acc0[i];
        float v1 = acc1[i];
        if (EPI == 1) { v0 = sigm(v0 + bv);  v1 = sigm(v1 + bv); }
        if (EPI == 2) { v0 = tanh_fast(v0);  v1 = tanh_fast(v1); }
        C[(size_t)row0 * N + col]        = v0;
        C[(size_t)(row0 + 16) * N + col] = v1;
    }
}

// ---------------------------------------------------------------------------
// LSTM pointwise cell update (faithful to reference: gates already sigmoided,
// then sigmoid/tanh applied AGAIN to the splits).
// ---------------------------------------------------------------------------
__global__ __launch_bounds__(256)
void lstm_pointwise(const float* __restrict__ gates, const float* __restrict__ c_tm1,
                    float* __restrict__ h_out, float* __restrict__ c_out)
{
    const int idx = blockIdx.x * 256 + threadIdx.x;   // 0 .. 128*1024-1
    const int b = idx >> 10;
    const int j = idx & 1023;
    const float* gr = gates + (size_t)b * (4 * NHID);
    const float i_t = gr[j];
    const float f_t = gr[j + NHID];
    const float o_t = gr[j + 2 * NHID];
    const float g_t = gr[j + 3 * NHID];
    const float c  = sigm(f_t) * c_tm1[idx] + sigm(i_t) * tanh_fast(g_t);
    const float hv = sigm(o_t) * tanh_fast(c);
    c_out[idx] = c;
    h_out[idx] = hv;
}

// ---------------------------------------------------------------------------
// Fused attention: scores = ctx[b] @ target[b]; softmax over L; z = a @ ctx[b].
// One workgroup (256 thr / 8 waves) per batch row; exactly two coalesced
// passes over that row's 2 MB slab of ctx (the op's roofline bottleneck).
// ---------------------------------------------------------------------------
__global__ __launch_bounds__(256)
void attention(const float* __restrict__ ctx, const float* __restrict__ tgt_g,
               float* __restrict__ z)
{
    __shared__ float tgt[DCTX];
    __shared__ float sc[LCTX];
    __shared__ float red[256];

    const int b    = blockIdx.x;
    const int tid  = threadIdx.x;
    const int lane = tid & 31;
    const int wave = tid >> 5;
    const float* cb = ctx + (size_t)b * LCTX * DCTX;

    #pragma unroll
    for (int i = 0; i < 4; ++i)
        tgt[tid + 256 * i] = tgt_g[(size_t)b * DCTX + tid + 256 * i];
    __syncthreads();

    // Pass 1: one wave per l, lanes stride D (contiguous 128B lines per wave)
    for (int l = wave; l < LCTX; l += 8) {
        const float* row = cb + (size_t)l * DCTX;
        float p = 0.0f;
        #pragma unroll 4
        for (int d = lane; d < DCTX; d += 32) p += row[d] * tgt[d];
        #pragma unroll
        for (int off = 16; off; off >>= 1) p += __shfl_xor(p, off, 32);
        if (lane == 0) sc[l] = p;
    }
    __syncthreads();

    // Softmax over 512 scores
    red[tid] = fmaxf(sc[tid], sc[tid + 256]);
    __syncthreads();
    for (int s = 128; s > 0; s >>= 1) {
        if (tid < s) red[tid] = fmaxf(red[tid], red[tid + s]);
        __syncthreads();
    }
    const float mx = red[0];
    __syncthreads();
    const float e0 = __expf(sc[tid] - mx);
    const float e1 = __expf(sc[tid + 256] - mx);
    red[tid] = e0 + e1;
    __syncthreads();
    for (int s = 128; s > 0; s >>= 1) {
        if (tid < s) red[tid] += red[tid + s];
        __syncthreads();
    }
    const float inv = __builtin_amdgcn_rcpf(red[0]);
    sc[tid]       = e0 * inv;
    sc[tid + 256] = e1 * inv;
    __syncthreads();

    // Pass 2: z[d] = sum_l a[l] * ctx[b,l,d]; threads span D (fully coalesced)
    float z0 = 0.f, z1 = 0.f, z2 = 0.f, z3 = 0.f;
    for (int l = 0; l < LCTX; ++l) {
        const float* row = cb + (size_t)l * DCTX;
        const float a = sc[l];
        z0 += a * row[tid];
        z1 += a * row[tid + 256];
        z2 += a * row[tid + 512];
        z3 += a * row[tid + 768];
    }
    float* zb = z + (size_t)b * DCTX;
    zb[tid]       = z0;
    zb[tid + 256] = z1;
    zb[tid + 512] = z2;
    zb[tid + 768] = z3;
}

// ---------------------------------------------------------------------------
extern "C" void kernel_launch(void* const* d_in, const int* in_sizes, int n_in,
                              void* d_out, int out_size, void* d_ws, size_t ws_size,
                              hipStream_t stream)
{
    const float* x       = (const float*)d_in[0];   // [128, 512]
    const float* out_tm1 = (const float*)d_in[1];   // [128, 1024]
    // d_in[2] = h_tm1 (unused by the reference)
    const float* c_tm1   = (const float*)d_in[3];   // [128, 1024]
    const float* ctx     = (const float*)d_in[4];   // [128, 512, 1024]
    const float* Wg      = (const float*)d_in[5];   // [1536, 4096]
    const float* bg      = (const float*)d_in[6];   // [4096]
    const float* Wt      = (const float*)d_in[7];   // [1024, 1024]
    const float* Wo      = (const float*)d_in[8];   // [2048, 1024]

    float* out   = (float*)d_out;                 // output_t [128,1024]
    float* h_out = out + B_SZ * NHID;             // h_t
    float* c_out = out + 2 * B_SZ * NHID;         // c_t

    float* ws     = (float*)d_ws;
    float* gates  = ws;                           // 128*4096
    float* target = gates + B_SZ * 4 * NHID;      // 128*1024
    float* zbuf   = target + B_SZ * NHID;         // 128*1024

    // 1) gates = sigmoid([x | out_tm1] @ Wg + bg)   M=128 N=4096 K=1536
    gemm_wmma<1><<<dim3(2, 64), 256, 0, stream>>>(
        x, NIN, NIN, out_tm1, NHID, Wg, bg, gates, 4 * NHID, NIN + NHID);

    // 2) c_t, h_t (written straight into their d_out slots)
    lstm_pointwise<<<(B_SZ * NHID) / 256, 256, 0, stream>>>(gates, c_tm1, h_out, c_out);

    // 3) target = h_t @ Wt                           M=128 N=1024 K=1024
    gemm_wmma<0><<<dim3(2, 16), 256, 0, stream>>>(
        h_out, NHID, NHID, h_out, NHID, Wt, nullptr, target, NHID, NHID);

    // 4) attention: scores -> softmax -> z (two coalesced passes over ctx)
    attention<<<B_SZ, 256, 0, stream>>>(ctx, target, zbuf);

    // 5) output_t = tanh([h_t | z] @ Wo)             M=128 N=1024 K=2048
    gemm_wmma<2><<<dim3(2, 16), 256, 0, stream>>>(
        h_out, NHID, NHID, zbuf, NHID, Wo, nullptr, out, NHID, NHID + DCTX);
}
